// GaussianImage_Cholesky_11613591568425
// MI455X (gfx1250) — matrix operand order repaired and checked by reading, not compiled, not run on hardware
//
#include <hip/hip_runtime.h>

// GaussianImage Cholesky render for gfx1250 (MI455X, wave32, WMMA).
//
// gi_setup writes d_ws in planar (SoA) form, 6 planes of 512 f32 per frame:
//   [cx | cy | 0.5*ca | cb | 0.5*cc | pad]
// and 4 plane-major f16 color planes premultiplied by sigmoid(opacity):
//   [opa*R | opa*G | opa*B | zeros]
// gi_render: each wave renders 16 consecutive pixels of one row; K-loop over
// 512 gaussians in chunks of 32 feeding v_wmma_f32_16x16x32_f16.
//   - A params: per group-of-8 gaussians, 5 v8f plane reads -> ds_load_b128
//     pairs with immediate offsets off one base pointer
//   - alphas fully branchless: arg = sigma>=0 ? sigma : +inf; a = exp(-arg)
//     (v_exp_f32(-inf)==0 implements the where() gate, incl. NaN -> 0);
//     opacity is pre-folded into the B colors, so alpha needs no extra mul
//   - B tile: one contiguous 32B LDS read from plane min(col,3) (zero plane
//     kills columns 3..15)
//   - output: clamp-modifier clip + 2x global_store_b128 per lane (col<3)

typedef __attribute__((ext_vector_type(16))) _Float16 v16h;
typedef __attribute__((ext_vector_type(8)))  float    v8f;

#define T_FRAMES 2
#define NGAUSS   512
#define IMG_H    256
#define IMG_W    256
#define WAVES_PER_FRAME (IMG_H * (IMG_W / 16))    // 4096
#define BLOCKS ((T_FRAMES * WAVES_PER_FRAME) / 8) // 1024 blocks of 8 waves

#define NPLANES 6                                  // 5 used + 1 pad plane
#define FRAME_FLOATS (NPLANES * NGAUSS)            // 3072 f32 = 12 KB / frame
#define WS_PARAMS_FLOATS (T_FRAMES * FRAME_FLOATS) // 24 KB

__global__ __launch_bounds__(NGAUSS) void gi_setup(
    const float* __restrict__ xyz, const float* __restrict__ chol,
    const float* __restrict__ opac, const float* __restrict__ feat,
    float* __restrict__ params, _Float16* __restrict__ colors)
{
    const int t  = blockIdx.x;
    const int n  = threadIdx.x;
    const int tn = t * NGAUSS + n;

    float mx = tanhf(xyz[tn * 2 + 0]);
    float my = tanhf(xyz[tn * 2 + 1]);
    float cx = 0.5f * IMG_W * (mx + 1.0f);
    float cy = 0.5f * IMG_H * (my + 1.0f);

    float l0 = chol[tn * 3 + 0] + 0.5f;      // CHOL_BOUND = [0.5, 0, 0.5]
    float l1 = chol[tn * 3 + 1];
    float l2 = chol[tn * 3 + 2] + 0.5f;
    float sxx = l0 * l0;
    float sxy = l0 * l1;
    float syy = l1 * l1 + l2 * l2;
    float det = sxx * syy - sxy * sxy;
    float inv = 1.0f / det;

    float* P = params + (size_t)t * FRAME_FLOATS;  // planar per-frame layout
    P[0 * NGAUSS + n] = cx;
    P[1 * NGAUSS + n] = cy;
    P[2 * NGAUSS + n] = 0.5f * syy * inv;   // 0.5*ca (sign test unaffected)
    P[3 * NGAUSS + n] = -sxy * inv;         // cb
    P[4 * NGAUSS + n] = 0.5f * sxx * inv;   // 0.5*cc
    P[5 * NGAUSS + n] = 0.0f;               // pad plane (keeps staging exact)

    if (t == 0) {
        float opa = 1.0f / (1.0f + __expf(-opac[n]));   // sigmoid(opacity)
        #pragma unroll
        for (int c = 0; c < 3; ++c)
            colors[c * NGAUSS + n] = (_Float16)(
                opa * (1.0f / (1.0f + __expf(-feat[n * 3 + c]))));
        colors[3 * NGAUSS + n] = (_Float16)0.0f;   // zero plane for cols 3..15
    }
}

// Compute 8 alphas for gaussians [base, base+8) into a[e0..e0+7].
// sp is the planar LDS base offset by kbaseA; planes are NGAUSS apart, so all
// five v8f reads are immediate-offset ds_load_b128 pairs off one address.
__device__ __forceinline__ void gi_alpha8(const float* __restrict__ spA,
                                          int base, float px, float py,
                                          v16h& a, int e0)
{
    v8f cx = *(const v8f*)(spA + 0 * NGAUSS + base);
    v8f cy = *(const v8f*)(spA + 1 * NGAUSS + base);
    v8f ca = *(const v8f*)(spA + 2 * NGAUSS + base);
    v8f cb = *(const v8f*)(spA + 3 * NGAUSS + base);
    v8f cc = *(const v8f*)(spA + 4 * NGAUSS + base);
    #pragma unroll
    for (int i = 0; i < 8; ++i) {
        float dx = cx[i] - px;
        float dy = cy[i] - py;
        float t1 = fmaf(cb[i], dy, ca[i] * dx);     // cb*dy + ca2*dx
        float w  = (cc[i] * dy) * dy;               // cc2*dy*dy
        float sg = fmaf(dx, t1, w);                 // full sigma
        float arg = (sg >= 0.0f) ? sg : __builtin_inff();  // cndmask only
        a[e0 + i] = (_Float16)__expf(-arg);         // exp(-inf)=0 gates
    }
}

__global__ __launch_bounds__(256) void gi_render(
    const float* __restrict__ params, const _Float16* __restrict__ colors,
    float* __restrict__ out)
{
    __shared__ float    sp[FRAME_FLOATS];   // 12 KB planar params
    __shared__ _Float16 sc[4 * NGAUSS];     //  4 KB premult. color planes

    const int tid = threadIdx.x;
    const int gw0 = blockIdx.x * 8;                  // first wave of block
    const int t   = gw0 / WAVES_PER_FRAME;           // 512 blocks per frame

    // Stage this frame's planes (float4 traffic) + color planes into LDS.
    {
        const float4* src = (const float4*)(params + (size_t)t * FRAME_FLOATS);
        float4*       dst = (float4*)sp;
        #pragma unroll
        for (int i = 0; i < (FRAME_FLOATS / 4) / 256; ++i)   // 3 per thread
            dst[tid + i * 256] = src[tid + i * 256];
        ((float4*)sc)[tid] = ((const float4*)colors)[tid];   // 256*16B = 4 KB
    }
    __syncthreads();

    const int wave = tid >> 5;
    const int lane = tid & 31;
    const int half = lane >> 4;     // 0 = lanes 0-15, 1 = lanes 16-31
    const int col  = lane & 15;     // A: M (pixel); B/D: N (channel)

    const int gw  = gw0 + wave;
    const int rem = gw % WAVES_PER_FRAME;
    const int y   = rem >> 4;                 // pixel row
    const int x0  = (rem & 15) << 4;          // 16-pixel strip start

    const float px = (float)(x0 + col);       // this lane's A-row pixel
    const float py = (float)y;
    const int kbaseA = half * 16;             // 16-bit A 16x32 lane K-mapping
    const int kbaseB = half * 16;             // 16-bit B 32x16 lane K-mapping
    const int ccol   = (col < 3) ? col : 3;   // plane 3 is all zeros

    const float*    spA = sp + kbaseA;
    const _Float16* pb  = sc + ccol * NGAUSS + kbaseB;

    v8f acc = {};
    for (int kb = 0; kb < NGAUSS; kb += 32) {
        // ---- A operand: 16 alphas, contiguous k order (same as B) ----
        // elements 0..7  -> gaussians kb + kbaseA + 0..7
        // elements 8..15 -> gaussians kb + kbaseA + 8..15
        v16h a;
        gi_alpha8(spA, kb,      px, py, a, 0);
        gi_alpha8(spA, kb + 8,  px, py, a, 8);

        // ---- B operand: 16 contiguous f16 from this lane's color plane ----
        v16h b = *(const v16h*)(pb + kb);     // 32B aligned -> 2x ds_load_b128

        // D(16x16 f32) += A(16x32 f16 alphas) x B(32x16 f16 opa*colors)
        acc = __builtin_amdgcn_wmma_f32_16x16x32_f16(
            /*neg_a=*/false, a, /*neg_b=*/false, b,
            /*c_mod=*/(short)0, acc, /*reuse_a=*/false, /*reuse_b=*/false);
    }

    // D layout: VGPR r -> pixel M = r + half*8, channel N = col.
    if (col < 3) {
        float* o = out + (((size_t)t * 3 + col) * IMG_H + y) * IMG_W
                       + x0 + half * 8;
        float4 v0, v1;
        v0.x = fminf(fmaxf(acc[0], 0.0f), 1.0f);   // v_max_num_f32 ... clamp
        v0.y = fminf(fmaxf(acc[1], 0.0f), 1.0f);
        v0.z = fminf(fmaxf(acc[2], 0.0f), 1.0f);
        v0.w = fminf(fmaxf(acc[3], 0.0f), 1.0f);
        v1.x = fminf(fmaxf(acc[4], 0.0f), 1.0f);
        v1.y = fminf(fmaxf(acc[5], 0.0f), 1.0f);
        v1.z = fminf(fmaxf(acc[6], 0.0f), 1.0f);
        v1.w = fminf(fmaxf(acc[7], 0.0f), 1.0f);
        *(float4*)(o + 0) = v0;               // 2x global_store_b128
        *(float4*)(o + 4) = v1;
    }
}

extern "C" void kernel_launch(void* const* d_in, const int* in_sizes, int n_in,
                              void* d_out, int out_size, void* d_ws, size_t ws_size,
                              hipStream_t stream) {
    const float* xyz  = (const float*)d_in[0];   // (T, N, 2) f32
    const float* chol = (const float*)d_in[1];   // (T, N, 3) f32
    const float* opac = (const float*)d_in[2];   // (N, 1) f32
    const float* feat = (const float*)d_in[3];   // (N, 3) f32

    float*    params = (float*)d_ws;                       // 24 KB planar
    _Float16* colsf16 =
        (_Float16*)((char*)d_ws + (size_t)WS_PARAMS_FLOATS * sizeof(float));

    gi_setup<<<dim3(T_FRAMES), dim3(NGAUSS), 0, stream>>>(
        xyz, chol, opac, feat, params, colsf16);
    gi_render<<<dim3(BLOCKS), dim3(256), 0, stream>>>(
        params, colsf16, (float*)d_out);
}
